// DiffusionTestModel_16243566313753
// MI455X (gfx1250) — matrix operand order repaired and checked by reading, not compiled, not run on hardware
//
#include <hip/hip_runtime.h>
#include <hip/hip_bf16.h>

#define Nn   4096
#define Ee   32768
#define EMBd 8

typedef __attribute__((ext_vector_type(16))) __bf16 v16bf;
typedef __attribute__((ext_vector_type(8)))  float  v8f;

__device__ __forceinline__ unsigned int f2bf(float f) {
    unsigned int x = __float_as_uint(f);
    unsigned int r = x + 0x7FFFu + ((x >> 16) & 1u);   // round-to-nearest-even
    return r >> 16;
}

__device__ __forceinline__ float fast_tanh(float x) {
    float ax = fabsf(x);
    float e  = __expf(-2.0f * ax);
    float r  = (1.0f - e) / (1.0f + e);
    return copysignf(r, x);
}

// ---------------- small utility kernels ----------------

__global__ void k_init(float* __restrict__ p, int n, float v) {
    int i = blockIdx.x * blockDim.x + threadIdx.x;
    if (i < n) p[i] = v;
}

__global__ void k_deg(const int* __restrict__ col, const float* __restrict__ w,
                      float* __restrict__ deg, int E) {
    int e = blockIdx.x * blockDim.x + threadIdx.x;
    if (e < E) atomicAdd(&deg[col[e]], w[e]);
}

__global__ void k_dis(const float* __restrict__ deg, float* __restrict__ dis, int n) {
    int i = blockIdx.x * blockDim.x + threadIdx.x;
    if (i < n) {
        float d = deg[i];
        dis[i] = (d > 0.0f) ? rsqrtf(d) : 0.0f;
    }
}

// f32 -> bf16 conversion, 8 elements per thread (float4 x2 in, uint4 out)
__global__ void k_cvt(const float* __restrict__ src, unsigned int* __restrict__ dst, int n8) {
    int i = blockIdx.x * blockDim.x + threadIdx.x;
    if (i < n8) {
        const float4* s = reinterpret_cast<const float4*>(src) + (size_t)i * 2;
        float4 a = s[0], b = s[1];
        uint4 o;
        o.x = f2bf(a.x) | (f2bf(a.y) << 16);
        o.y = f2bf(a.z) | (f2bf(a.w) << 16);
        o.z = f2bf(b.x) | (f2bf(b.y) << 16);
        o.w = f2bf(b.z) | (f2bf(b.w) << 16);
        reinterpret_cast<uint4*>(dst)[i] = o;
    }
}

// one block per (edge or self-loop); 256 threads sweep the 4096-wide row
__global__ __launch_bounds__(256) void k_scatter(const int* __restrict__ ei,
                                                 const float* __restrict__ w,
                                                 const float* __restrict__ dis,
                                                 const float* __restrict__ h,
                                                 float* __restrict__ agg) {
    int edge = blockIdx.x;
    int r, c; float nrm;
    if (edge < Ee) {
        r = ei[edge];            // row
        c = ei[Ee + edge];       // col
        nrm = dis[r] * w[edge] * dis[c];
    } else {
        r = c = edge - Ee;       // self loop, weight 1
        float d = dis[r];
        nrm = d * d;
    }
    const float* hr = h + (size_t)r * Nn;
    float* ac = agg + (size_t)c * Nn;
    for (int j = threadIdx.x; j < Nn; j += blockDim.x)
        atomicAdd(&ac[j], nrm * hr[j]);
}

// H1 = tanh(agg + gcn_b), written directly as bf16 (8 elements / thread)
__global__ void k_h1(const float* __restrict__ agg, const float* __restrict__ b,
                     unsigned int* __restrict__ h1b, int n8) {
    int i = blockIdx.x * blockDim.x + threadIdx.x;
    if (i < n8) {
        int colj = (i * 8) & (Nn - 1);           // 8 | Nn, so cols are contiguous
        const float4* s  = reinterpret_cast<const float4*>(agg) + (size_t)i * 2;
        const float4* bb = reinterpret_cast<const float4*>(b + colj);
        float4 a0 = s[0], a1 = s[1];
        float4 b0 = bb[0], b1 = bb[1];
        float t0 = fast_tanh(a0.x + b0.x), t1 = fast_tanh(a0.y + b0.y);
        float t2 = fast_tanh(a0.z + b0.z), t3 = fast_tanh(a0.w + b0.w);
        float t4 = fast_tanh(a1.x + b1.x), t5 = fast_tanh(a1.y + b1.y);
        float t6 = fast_tanh(a1.z + b1.z), t7 = fast_tanh(a1.w + b1.w);
        uint4 o;
        o.x = f2bf(t0) | (f2bf(t1) << 16);
        o.y = f2bf(t2) | (f2bf(t3) << 16);
        o.z = f2bf(t4) | (f2bf(t5) << 16);
        o.w = f2bf(t6) | (f2bf(t7) << 16);
        reinterpret_cast<uint4*>(h1b)[i] = o;
    }
}

__global__ void k_final(const float* __restrict__ Facc, const float* __restrict__ emb,
                        const float* __restrict__ wf_w, const float* __restrict__ wf_b,
                        float* __restrict__ out) {
    int e = blockIdx.x * blockDim.x + threadIdx.x;
    if (e < Ee) {
        float acc = Facc[e] + wf_b[0];
        #pragma unroll
        for (int k = 0; k < EMBd; ++k)
            acc += emb[e * EMBd + k] * wf_w[Nn + k];
        out[e] = acc;
    }
}

// ---------------- WMMA bf16 GEMM core ----------------
// C[m,n] = sum_k A[m,k] * Bt[n,k]   (A, Bt row-major bf16)
// A fragment (16x32, lane l, half hi): elems 0..7 = K[k0+8*hi ..], elems 8..15 = K[k0+16+8*hi ..]
// B fragment (32x16): column n = base+l, elems 0..7 = K[k0+8*hi ..], elems 8..15 = K[k0+16+8*hi ..]

union FragU { v16bf v; uint4 q[2]; };

__device__ __forceinline__ v16bf load_a_frag(const unsigned short* __restrict__ row, int k0, int hi) {
    FragU t;
    t.q[0] = *reinterpret_cast<const uint4*>(row + k0 + hi * 8);
    t.q[1] = *reinterpret_cast<const uint4*>(row + k0 + 16 + hi * 8);
    return t.v;
}

__device__ __forceinline__ v16bf load_b_frag(const unsigned short* __restrict__ row, int k0, int hi) {
    FragU t;
    t.q[0] = *reinterpret_cast<const uint4*>(row + k0 + hi * 8);
    t.q[1] = *reinterpret_cast<const uint4*>(row + k0 + 16 + hi * 8);
    return t.v;
}

__device__ __forceinline__ void wmma_kloop(const unsigned short* __restrict__ arow0,
                                           const unsigned short* __restrict__ arow1,
                                           const unsigned short* __restrict__ arow2,
                                           const unsigned short* __restrict__ arow3,
                                           const unsigned short* __restrict__ brow0,
                                           const unsigned short* __restrict__ brow1,
                                           int K, int hi, v8f c[4][2]) {
    const unsigned short* arow[4] = {arow0, arow1, arow2, arow3};
    const unsigned short* brow[2] = {brow0, brow1};
    for (int k0 = 0; k0 < K; k0 += 32) {
        if (k0 + 32 < K) {                       // global_prefetch_b8 for next K chunk
            __builtin_prefetch(arow[0] + k0 + 32, 0, 3);
            __builtin_prefetch(brow[0] + k0 + 32, 0, 3);
        }
        v16bf a[4], b[2];
        #pragma unroll
        for (int mi = 0; mi < 4; ++mi) a[mi] = load_a_frag(arow[mi], k0, hi);
        #pragma unroll
        for (int ni = 0; ni < 2; ++ni) b[ni] = load_b_frag(brow[ni], k0, hi);
        #pragma unroll
        for (int mi = 0; mi < 4; ++mi)
            #pragma unroll
            for (int ni = 0; ni < 2; ++ni)
                c[mi][ni] = __builtin_amdgcn_wmma_f32_16x16x32_bf16(
                    false, a[mi], false, b[ni], (short)0, c[mi][ni], false, false);
    }
}

// GEMM1: C = A @ Bt^T, plain f32 store
__global__ __launch_bounds__(256) void k_gemm1(const unsigned short* __restrict__ A,
                                               const unsigned short* __restrict__ Bt,
                                               float* __restrict__ C,
                                               int K, int lda, int ldb, int ldc) {
    int tid  = threadIdx.x;
    int lane = tid & 31;
    int wave = tid >> 5;
    int l  = lane & 15;
    int hi = (lane >> 4) & 1;
    int mBase = blockIdx.x * 128 + (wave >> 2) * 64;
    int nBase = blockIdx.y * 128 + (wave & 3) * 32;

    v8f c[4][2];
    #pragma unroll
    for (int mi = 0; mi < 4; ++mi)
        #pragma unroll
        for (int ni = 0; ni < 2; ++ni)
            #pragma unroll
            for (int r = 0; r < 8; ++r) c[mi][ni][r] = 0.0f;

    wmma_kloop(A + (size_t)(mBase + 0 * 16 + l) * lda,
               A + (size_t)(mBase + 1 * 16 + l) * lda,
               A + (size_t)(mBase + 2 * 16 + l) * lda,
               A + (size_t)(mBase + 3 * 16 + l) * lda,
               Bt + (size_t)(nBase + 0 * 16 + l) * ldb,
               Bt + (size_t)(nBase + 1 * 16 + l) * ldb,
               K, hi, c);

    #pragma unroll
    for (int mi = 0; mi < 4; ++mi)
        #pragma unroll
        for (int ni = 0; ni < 2; ++ni) {
            int colj = nBase + ni * 16 + l;
            #pragma unroll
            for (int r = 0; r < 8; ++r) {
                int rowr = mBase + mi * 16 + r + 8 * hi;
                C[(size_t)rowr * ldc + colj] = c[mi][ni][r];
            }
        }
}

// GEMM2 fused: F[e] += sum_m wf_w[m] * tanh(C[m,e] + wq_b[e])
__global__ __launch_bounds__(256) void k_gemm2(const unsigned short* __restrict__ A,
                                               const unsigned short* __restrict__ Bt,
                                               const float* __restrict__ wq_b,
                                               const float* __restrict__ wf_w,
                                               float* __restrict__ F,
                                               int K, int lda, int ldb) {
    int tid  = threadIdx.x;
    int lane = tid & 31;
    int wave = tid >> 5;
    int l  = lane & 15;
    int hi = (lane >> 4) & 1;
    int mBase = blockIdx.x * 128 + (wave >> 2) * 64;   // x fastest: M tiles share Bt strip in L2
    int nBase = blockIdx.y * 128 + (wave & 3) * 32;

    v8f c[4][2];
    #pragma unroll
    for (int mi = 0; mi < 4; ++mi)
        #pragma unroll
        for (int ni = 0; ni < 2; ++ni)
            #pragma unroll
            for (int r = 0; r < 8; ++r) c[mi][ni][r] = 0.0f;

    wmma_kloop(A + (size_t)(mBase + 0 * 16 + l) * lda,
               A + (size_t)(mBase + 1 * 16 + l) * lda,
               A + (size_t)(mBase + 2 * 16 + l) * lda,
               A + (size_t)(mBase + 3 * 16 + l) * lda,
               Bt + (size_t)(nBase + 0 * 16 + l) * ldb,
               Bt + (size_t)(nBase + 1 * 16 + l) * ldb,
               K, hi, c);

    #pragma unroll
    for (int ni = 0; ni < 2; ++ni) {
        int e = nBase + ni * 16 + l;
        float qb = wq_b[e];
        float acc = 0.0f;
        #pragma unroll
        for (int mi = 0; mi < 4; ++mi) {
            #pragma unroll
            for (int r = 0; r < 8; ++r) {
                int rowr = mBase + mi * 16 + r + 8 * hi;
                acc += wf_w[rowr] * fast_tanh(c[mi][ni][r] + qb);
            }
        }
        acc += __shfl_xor(acc, 16, 32);   // combine the two M-halves (same e)
        if (hi == 0) atomicAdd(&F[e], acc);
    }
}

// ---------------- launch ----------------

extern "C" void kernel_launch(void* const* d_in, const int* in_sizes, int n_in,
                              void* d_out, int out_size, void* d_ws, size_t ws_size,
                              hipStream_t stream) {
    const float* x     = (const float*)d_in[0];
    const int*   ei    = (const int*)  d_in[1];   // [2,E]: row at +0, col at +E
    const float* ew    = (const float*)d_in[2];
    const float* gcn_w = (const float*)d_in[3];
    const float* gcn_b = (const float*)d_in[4];
    const float* wq_w  = (const float*)d_in[5];
    const float* wq_b  = (const float*)d_in[6];
    const float* emb   = (const float*)d_in[7];
    const float* wf_w  = (const float*)d_in[8];
    const float* wf_b  = (const float*)d_in[9];
    float* out = (float*)d_out;

    const size_t NN = (size_t)Nn * Nn;
    char* base = (char*)d_ws;
    float*          deg  = (float*)(base);                       // N
    float*          dis  = (float*)(base + (1u << 14));          // 16KB
    float*          Facc = (float*)(base + (2u << 14));          // 32KB
    float*          h    = (float*)(base + (1u << 20));          // 1MB   : N*N f32 (64MB)
    float*          agg  = (float*)(base + 65 * (1u << 20));     // 65MB  : N*N f32 (64MB)
    unsigned int*   xb   = (unsigned int*)(base + 129 * (size_t)(1u << 20)); // 32MB bf16
    unsigned int*   gwb  = (unsigned int*)(base + 161 * (size_t)(1u << 20)); // 32MB bf16
    unsigned int*   h1b  = (unsigned int*)(base + 193 * (size_t)(1u << 20)); // 32MB bf16
    unsigned int*   wqb  = (unsigned int*)(base + 225 * (size_t)(1u << 20)); // 256MB bf16

    const int NT = 256;
    // degree (self loops contribute 1.0) and normalization
    k_init<<<(Nn + NT - 1) / NT, NT, 0, stream>>>(deg, Nn, 1.0f);
    k_init<<<(Ee + NT - 1) / NT, NT, 0, stream>>>(Facc, Ee, 0.0f);
    k_init<<<(int)((NN + NT - 1) / NT), NT, 0, stream>>>(agg, (int)NN, 0.0f);
    k_deg<<<(Ee + NT - 1) / NT, NT, 0, stream>>>(ei + Ee, ew, deg, Ee);
    k_dis<<<(Nn + NT - 1) / NT, NT, 0, stream>>>(deg, dis, Nn);

    // one-shot bf16 conversions (each element converted exactly once)
    {
        int n8 = (int)(NN / 8);
        k_cvt<<<(n8 + NT - 1) / NT, NT, 0, stream>>>(x,     xb,  n8);
        k_cvt<<<(n8 + NT - 1) / NT, NT, 0, stream>>>(gcn_w, gwb, n8);
        int w8 = (int)(((size_t)Ee * Nn) / 8);
        k_cvt<<<(w8 + NT - 1) / NT, NT, 0, stream>>>(wq_w,  wqb, w8);
    }

    // h = x @ gcn_w.T   (bf16 WMMA, f32 accumulate)
    k_gemm1<<<dim3(Nn / 128, Nn / 128), NT, 0, stream>>>(
        (const unsigned short*)xb, (const unsigned short*)gwb, h, Nn, Nn, Nn, Nn);

    // agg[col] += norm * h[row]
    k_scatter<<<Ee + Nn, NT, 0, stream>>>(ei, ew, dis, h, agg);

    // H1 = tanh(agg + gcn_b) -> bf16
    {
        int n8 = (int)(NN / 8);
        k_h1<<<(n8 + NT - 1) / NT, NT, 0, stream>>>(agg, gcn_b, h1b, n8);
    }

    // fused: F[e] = sum_n wf_w[n]*tanh((H1 @ wq_w.T)[n,e] + wq_b[e])
    k_gemm2<<<dim3(Nn / 128, Ee / 128), NT, 0, stream>>>(
        (const unsigned short*)h1b, (const unsigned short*)wqb, wq_b, wf_w, Facc, Nn, Nn, Nn);

    // tail: + emb @ wf_w[N:] + wf_b
    k_final<<<(Ee + NT - 1) / NT, NT, 0, stream>>>(Facc, emb, wf_w, wf_b, out);
}
